// S4Block_10110353015035
// MI455X (gfx1250) — matrix operand order repaired and checked by reading, not compiled, not run on hardware
//
#include <hip/hip_runtime.h>
#include <hip/hip_bf16.h>

// S4 block for MI455X (gfx1250, wave32).
// Pipeline: [cauchy+2 FFTs per channel] -> [FFT conv + D*u + exact GELU] ->
//           [WMMA f16 GEMM w/ f32 accum + bias + residual] -> [LayerNorm].
// Roofline: FFT path ~3 GFLOP / ~150 MB traffic -> bandwidth bound (~10us at
// 23.3 TB/s). GEMM 8.6 GFLOP -> trivial for WMMA; fuse epilogues to save HBM.

#define D_MODEL 512
#define NSTATE  64
#define LMAX    2048
#define LFFT    4096
#define BATCH   8
#define MROWS   (BATCH * LMAX) // 16384

typedef __attribute__((ext_vector_type(16))) _Float16 v16h;
typedef __attribute__((ext_vector_type(8)))  _Float16 v8h;
typedef __attribute__((ext_vector_type(8)))  float    v8f;

// ---------------- complex helpers ----------------
__device__ __forceinline__ float2 cadd(float2 a, float2 b){ return make_float2(a.x+b.x, a.y+b.y); }
__device__ __forceinline__ float2 csub(float2 a, float2 b){ return make_float2(a.x-b.x, a.y-b.y); }
__device__ __forceinline__ float2 cmul(float2 a, float2 b){ return make_float2(a.x*b.x - a.y*b.y, a.x*b.y + a.y*b.x); }
__device__ __forceinline__ float2 cdivc(float2 a, float2 b){
  float inv = 1.0f / (b.x*b.x + b.y*b.y);
  return make_float2((a.x*b.x + a.y*b.y)*inv, (a.y*b.x - a.x*b.y)*inv);
}
__device__ __forceinline__ float2 cinv(float2 b){
  float inv = 1.0f / (b.x*b.x + b.y*b.y);
  return make_float2(b.x*inv, -b.y*inv);
}

// ---------------- in-place radix-2 FFT in LDS ----------------
// sgn = -1: forward; sgn = +1: inverse (unscaled; caller applies 1/n).
__device__ void fft_inplace(float2* x, int n, int logn, float sgn, int tid, int nt)
{
  // bit-reversal permutation (pair touched only by its smaller index)
  for (int i = tid; i < n; i += nt) {
    int j = (int)(__brev((unsigned)i) >> (32 - logn));
    if (i < j) { float2 t = x[i]; x[i] = x[j]; x[j] = t; }
  }
  __syncthreads();
  for (int len = 2; len <= n; len <<= 1) {
    const int half = len >> 1;
    const float base = sgn * 6.283185307179586f / (float)len;
    for (int i = tid; i < (n >> 1); i += nt) {
      int blk = i / half;
      int k   = i - blk * half;
      int i0  = blk * len + k;
      int i1  = i0 + half;
      float sv, cv; __sincosf(base * (float)k, &sv, &cv);
      float2 w = make_float2(cv, sv);
      float2 a = x[i0];
      float2 b = cmul(w, x[i1]);
      x[i0] = cadd(a, b);
      x[i1] = csub(a, b);
    }
    __syncthreads();
  }
}

// ---------------- kernel 0: W -> f16 ----------------
__global__ void __launch_bounds__(256) convert_w_f16(const float* __restrict__ W,
                                                     _Float16* __restrict__ Wh, int n)
{
  int i = blockIdx.x * 256 + threadIdx.x;
  if (i < n) Wh[i] = (_Float16)W[i];
}

// ---------------- kernel 1: Cauchy generating fn + iFFT(2048) + FFT(4096) ----------------
// One block per channel d. Produces KF[d, 0..4095]: full spectrum of padded K.
__global__ void __launch_bounds__(256) s4_build_kernel_spectrum(
    const float* __restrict__ lam, const float* __restrict__ p, const float* __restrict__ q,
    const float* __restrict__ Bm,  const float* __restrict__ Ct,
    const float* __restrict__ log_step, float2* __restrict__ KF)
{
  __shared__ float2 buf[LFFT];
  __shared__ float2 lamS[NSTATE], v0S[NSTATE], v1S[NSTATE], v2S[NSTATE], v3S[NSTATE];
  const int d = blockIdx.x, tid = threadIdx.x;

  if (tid < NSTATE) {
    int n = tid;
    float2 lm = make_float2(lam[2*n],   lam[2*n+1]);
    float2 pv = make_float2(p[2*n],     p[2*n+1]);
    float2 qc = make_float2(q[2*n],    -q[2*n+1]);                       // conj(q)
    float2 Bv = make_float2(Bm[(d*NSTATE+n)*2],  Bm[(d*NSTATE+n)*2+1]);
    float2 Cc = make_float2(Ct[(d*NSTATE+n)*2], -Ct[(d*NSTATE+n)*2+1]);  // conj(Ct)
    lamS[n] = lm;
    v0S[n] = cmul(Cc, Bv); v1S[n] = cmul(Cc, pv);
    v2S[n] = cmul(qc, Bv); v3S[n] = cmul(qc, pv);
  }
  __syncthreads();

  const float twos = 2.0f / __expf(log_step[d]);
  for (int l = tid; l < LMAX; l += 256) {
    float ang = 6.283185307179586f * (float)l / (float)LMAX;
    float s, c; __sincosf(ang, &s, &c);
    float2 onep = make_float2(1.0f + c,  s);
    float2 onem = make_float2(1.0f - c, -s);
    float2 f = cdivc(onem, onep); f.x *= twos; f.y *= twos;
    float2 k0 = make_float2(0.f, 0.f), k1 = k0, k2 = k0, k3 = k0;
    #pragma unroll 8
    for (int n = 0; n < NSTATE; ++n) {
      float2 r = cinv(csub(f, lamS[n]));
      k0 = cadd(k0, cmul(v0S[n], r));
      k1 = cadd(k1, cmul(v1S[n], r));
      k2 = cadd(k2, cmul(v2S[n], r));
      k3 = cadd(k3, cmul(v3S[n], r));
    }
    float2 pref  = cdivc(make_float2(2.f, 0.f), onep);
    float2 inner = csub(k0, cmul(k1, cdivc(k2, cadd(make_float2(1.f, 0.f), k3))));
    buf[l] = cmul(pref, inner);
  }
  __syncthreads();

  fft_inplace(buf, LMAX, 11, +1.0f, tid, 256);   // iFFT (unscaled)
  const float invL = 1.0f / (float)LMAX;
  for (int l = tid; l < LFFT; l += 256)          // K = Re(ifft)/L, zero-pad to 4096
    buf[l] = (l < LMAX) ? make_float2(buf[l].x * invL, 0.f) : make_float2(0.f, 0.f);
  __syncthreads();
  fft_inplace(buf, LFFT, 12, -1.0f, tid, 256);   // forward FFT of padded K
  for (int j = tid; j < LFFT; j += 256) KF[(size_t)d * LFFT + j] = buf[j];
}

// ---------------- kernel 2: FFT conv + D*u + exact GELU -> f16 activations ----------------
// One block per (d, b). 4096-pt FFT of zero-padded u, pointwise * KF, iFFT.
__global__ void __launch_bounds__(256) s4_fft_conv(
    const float* __restrict__ u, const float2* __restrict__ KF,
    const float* __restrict__ Dg, _Float16* __restrict__ gh)
{
  __shared__ float2 buf[LFFT];
  const int d = blockIdx.x, b = blockIdx.y, tid = threadIdx.x;

  for (int l = tid; l < LFFT; l += 256)
    buf[l] = (l < LMAX) ? make_float2(u[((size_t)b*LMAX + l)*D_MODEL + d], 0.f)
                        : make_float2(0.f, 0.f);
  __syncthreads();
  fft_inplace(buf, LFFT, 12, -1.0f, tid, 256);
  for (int j = tid; j < LFFT; j += 256) buf[j] = cmul(buf[j], KF[(size_t)d*LFFT + j]);
  __syncthreads();
  fft_inplace(buf, LFFT, 12, +1.0f, tid, 256);

  const float invN = 1.0f / (float)LFFT;
  const float Dd = Dg[d];
  for (int l = tid; l < LMAX; l += 256) {
    float uv = u[((size_t)b*LMAX + l)*D_MODEL + d];
    float y  = buf[l].x * invN + Dd * uv;
    float g  = 0.5f * y * (1.0f + erff(y * 0.70710678118654752f));  // exact GELU
    gh[((size_t)b*LMAX + l)*D_MODEL + d] = (_Float16)g;
  }
}

// ---------------- kernel 3: Y = A @ W^T + bias + u via v_wmma_f32_16x16x32_f16 ----------------
// 8 waves/block; each wave owns one 16x16 tile. Block shares one M-tile
// (A fragments hit in cache across its 8 N-tiles). Fragment packing follows
// ISA 7.12.2 layout for f32 C/D (16x16); 16-bit A and B use contiguous k = 16*hl + s instead.
__global__ void __launch_bounds__(256) s4_linear_wmma(
    const _Float16* __restrict__ A,   // (M, K)
    const _Float16* __restrict__ Wh,  // (N, K): row j = output channel j
    const float* __restrict__ bias, const float* __restrict__ u,
    float* __restrict__ Y)            // (M, N)
{
  const int tid  = threadIdx.x;
  const int wave = tid >> 5, lane = tid & 31;
  const int hl   = lane >> 4, lrow = lane & 15;
  const int mt   = blockIdx.x >> 2;
  const int nt   = ((blockIdx.x & 3) << 3) + wave;
  const int m0 = mt << 4, n0 = nt << 4;
  const int K = D_MODEL, Nn = D_MODEL;

  const _Float16* arow = A  + (size_t)(m0 + lrow) * K;
  const _Float16* brow = Wh + (size_t)(n0 + lrow) * K;
  v8f acc = {};
  for (int kk = 0; kk < K; kk += 32) {
    // A 16x32 f16: lane(hl,lrow) holds K in [hl*16, hl*16+16)
    v8h a_lo = *(const v8h*)(arow + kk + hl*16);
    v8h a_hi = *(const v8h*)(arow + kk + hl*16 + 8);
    v16h af;
    #pragma unroll
    for (int i = 0; i < 8; ++i) { af[i] = a_lo[i]; af[8+i] = a_hi[i]; }
    // B 32x16 f16: B[k,j] = W[j,k]; lane(hl,j) holds W[j, kk+hl*16 .. +15] (32B contig)
    v16h bf = *(const v16h*)(brow + kk + hl*16);
    acc = __builtin_amdgcn_wmma_f32_16x16x32_f16(false, af, false, bf,
                                                 (short)0, acc, false, false);
  }
  // D 16x16 f32: VGPR r -> row r + 8*hl, col = lane&15. Fuse bias + residual.
  const int col = n0 + lrow;
  const float bv = bias[col];
  #pragma unroll
  for (int r = 0; r < 8; ++r) {
    int m = m0 + (hl << 3) + r;
    Y[(size_t)m*Nn + col] = acc[r] + bv + u[(size_t)m*Nn + col];
  }
}

// ---------------- kernel 4: LayerNorm over channels ----------------
__global__ void __launch_bounds__(256) s4_layernorm(
    const float* __restrict__ Y, const float* __restrict__ gamma,
    const float* __restrict__ beta, float* __restrict__ out)
{
  __shared__ float red[256];
  const int m = blockIdx.x, tid = threadIdx.x;
  float v0 = Y[(size_t)m*D_MODEL + tid];
  float v1 = Y[(size_t)m*D_MODEL + tid + 256];
  red[tid] = v0 + v1; __syncthreads();
  for (int s = 128; s > 0; s >>= 1) { if (tid < s) red[tid] += red[tid + s]; __syncthreads(); }
  const float mu = red[0] * (1.0f / D_MODEL);
  __syncthreads();
  float d0 = v0 - mu, d1 = v1 - mu;
  red[tid] = d0*d0 + d1*d1; __syncthreads();
  for (int s = 128; s > 0; s >>= 1) { if (tid < s) red[tid] += red[tid + s]; __syncthreads(); }
  const float rstd = rsqrtf(red[0] * (1.0f / D_MODEL) + 1e-5f);
  out[(size_t)m*D_MODEL + tid]       = d0 * rstd * gamma[tid]       + beta[tid];
  out[(size_t)m*D_MODEL + tid + 256] = d1 * rstd * gamma[tid + 256] + beta[tid + 256];
}

// ---------------- host launcher ----------------
extern "C" void kernel_launch(void* const* d_in, const int* in_sizes, int n_in,
                              void* d_out, int out_size, void* d_ws, size_t ws_size,
                              hipStream_t stream)
{
  const float* u   = (const float*)d_in[0];
  const float* lam = (const float*)d_in[1];
  const float* p   = (const float*)d_in[2];
  const float* q   = (const float*)d_in[3];
  const float* Bm  = (const float*)d_in[4];
  const float* Ct  = (const float*)d_in[5];
  const float* Dg  = (const float*)d_in[6];
  const float* ls  = (const float*)d_in[7];
  const float* W   = (const float*)d_in[8];
  const float* bb  = (const float*)d_in[9];
  const float* gm  = (const float*)d_in[10];
  const float* bt  = (const float*)d_in[11];

  char* ws = (char*)d_ws;
  float2*   KF = (float2*)(ws);                                   // 16 MB
  _Float16* gh = (_Float16*)(ws + (size_t)16*1024*1024);          // 16 MB
  _Float16* Wh = (_Float16*)(ws + (size_t)32*1024*1024);          // 0.5 MB
  float*    Y  = (float*)  (ws + (size_t)33*1024*1024);           // 32 MB

  convert_w_f16<<<dim3((D_MODEL*D_MODEL + 255)/256), 256, 0, stream>>>(W, Wh, D_MODEL*D_MODEL);
  s4_build_kernel_spectrum<<<dim3(D_MODEL), 256, 0, stream>>>(lam, p, q, Bm, Ct, ls, KF);
  s4_fft_conv<<<dim3(D_MODEL, BATCH), 256, 0, stream>>>(u, KF, Dg, gh);
  s4_linear_wmma<<<dim3((MROWS/16) * (D_MODEL/16) / 8), 256, 0, stream>>>(gh, Wh, bb, u, Y);
  s4_layernorm<<<dim3(MROWS), 256, 0, stream>>>(Y, gm, bt, (float*)d_out);
}
